// MultiScaleDHSM_37014028157381
// MI455X (gfx1250) — matrix operand order repaired and checked by reading, not compiled, not run on hardware
//
#include <hip/hip_runtime.h>
#include <hip/hip_bf16.h>
#include <math.h>

#define BB 2
#define SS 2048
#define HH 1024
#define VV 32000
#define MM (BB * SS)   // 4096 token rows

#define BSTRIDE 40     // LDS row stride in halves (64B rows would 4-way conflict)

typedef __attribute__((ext_vector_type(16))) __bf16 v16bf;
typedef __attribute__((ext_vector_type(8)))  __bf16 v8bf;
typedef __attribute__((ext_vector_type(8)))  float  v8f;

// ---------------------------------------------------------------- helpers

__device__ __forceinline__ unsigned short f32_to_bf16(float f) {
  unsigned u = __builtin_bit_cast(unsigned, f);
  u += 0x7FFFu + ((u >> 16) & 1u);          // round-to-nearest-even
  return (unsigned short)(u >> 16);
}

// Workgroup barrier that only drains DScnt (LDS ops). Unlike __syncthreads(),
// it does NOT wait for in-flight global loads into private VGPRs, so next-tile
// loads issued before the barrier keep flying through it.
__device__ __forceinline__ void lds_barrier() {
  asm volatile(
      "s_wait_dscnt 0x0\n"
      "s_barrier_signal -1\n"
      "s_barrier_wait -1"
      ::: "memory");
}

// A-fragment (16x32 bf16, M x K): lane holds row lane%16.
// Per ISA 7.12.2: element j <-> k = (j/8)*16 + (lane/16)*8 + (j%8)
// -> two contiguous 8-half (16B) chunks at k0+(lane/16)*8 and +16.
__device__ __forceinline__ v16bf load_a_frag(const unsigned short* __restrict__ A,
                                             int lda, int m0, int k0, int lane) {
  const int half = lane >> 4, lrow = lane & 15;
  const unsigned short* p = A + (size_t)(m0 + lrow) * lda + k0 + half * 8;
  union { v16bf v; v8bf h[2]; } u;
  u.h[0] = *(const v8bf*)(p);
  u.h[1] = *(const v8bf*)(p + 16);
  return u.v;
}

// B-fragment (32x16 bf16, K x N) sourced from an LDS-staged tile whose rows are
// weight rows (columns of B), stride BSTRIDE halves. lane holds column lane%16,
// k = (j/8)*16 + (lane/16)*8 + (j%8) -> two 8-half (16B) chunks, matching A.
__device__ __forceinline__ v16bf lds_b_frag(const unsigned short* Bs, int t, int lane) {
  const int half = lane >> 4, lrow = lane & 15;
  const unsigned short* p = Bs + (t * 16 + lrow) * BSTRIDE + half * 8;
  union { v16bf v; v8bf h[2]; } u;
  u.h[0] = *(const v8bf*)(p);
  u.h[1] = *(const v8bf*)(p + 16);
  return u.v;
}

__device__ __forceinline__ v8f wmma_bf16(v16bf a, v16bf b, v8f c) {
  return __builtin_amdgcn_wmma_f32_16x16x32_bf16(false, a, false, b, (short)0, c,
                                                 false, false);
}

// One k-step of the 32x64 wave tile: batch the 4 B-frag ds_loads, then 8 WMMAs.
__device__ __forceinline__ void mma_step(const unsigned short* bs, int lane,
                                         const v16bf& a0, const v16bf& a1,
                                         v8f acc[2][4]) {
  v16bf b0 = lds_b_frag(bs, 0, lane);
  v16bf b1 = lds_b_frag(bs, 1, lane);
  v16bf b2 = lds_b_frag(bs, 2, lane);
  v16bf b3 = lds_b_frag(bs, 3, lane);
  acc[0][0] = wmma_bf16(a0, b0, acc[0][0]);
  acc[1][0] = wmma_bf16(a1, b0, acc[1][0]);
  acc[0][1] = wmma_bf16(a0, b1, acc[0][1]);
  acc[1][1] = wmma_bf16(a1, b1, acc[1][1]);
  acc[0][2] = wmma_bf16(a0, b2, acc[0][2]);
  acc[1][2] = wmma_bf16(a1, b2, acc[1][2]);
  acc[0][3] = wmma_bf16(a0, b3, acc[0][3]);
  acc[1][3] = wmma_bf16(a1, b3, acc[1][3]);
}

// ---------------------------------------------------------------- small kernels

__global__ void cvt_kernel(const float* __restrict__ in,
                           unsigned short* __restrict__ out, size_t n) {
  size_t i = (size_t)blockIdx.x * blockDim.x + threadIdx.x;
  if (i < n) out[i] = f32_to_bf16(in[i]);
}

__global__ void dplus1_kernel(const float* __restrict__ D, float* __restrict__ out, int n) {
  int i = blockIdx.x * blockDim.x + threadIdx.x;
  if (i < n) out[i] = D[i] + 1.0f;
}

__global__ void embed_kernel(const int* __restrict__ xi, const float* __restrict__ E,
                             float* __restrict__ ef, unsigned short* __restrict__ eb,
                             size_t total) {
  size_t i = (size_t)blockIdx.x * blockDim.x + threadIdx.x;
  if (i >= total) return;
  size_t row = i / HH;
  int col = (int)(i % HH);
  float v = E[(size_t)xi[row] * HH + col];
  ef[i] = v;
  eb[i] = f32_to_bf16(v);
}

// Serial state-space scan: one thread per (batch, channel). Coalesced across n.
__global__ void scan_kernel(const float* __restrict__ U, const float* __restrict__ Alog,
                            unsigned short* __restrict__ Sb, int N) {
  int t = blockIdx.x * blockDim.x + threadIdx.x;
  if (t >= BB * N) return;
  int b = t / N, n = t % N;
  float a = expf(Alog[n]);
  float s = 0.0f;
  const float* up = U + (size_t)b * SS * N + n;
  unsigned short* sp = Sb + (size_t)b * SS * N + n;
  for (int i = 0; i < SS; ++i) {
    s = fmaf(a, s, up[(size_t)i * N]);
    sp[(size_t)i * N] = f32_to_bf16(s);
  }
}

// Row LayerNorm over H=1024 (f32 in, bf16 out). One 256-thread block per row.
__global__ void __launch_bounds__(256) ln_kernel(const float* __restrict__ X,
                                                 const float* __restrict__ gam,
                                                 const float* __restrict__ bet,
                                                 unsigned short* __restrict__ out) {
  __shared__ float sh1[256], sh2[256];
  const int row = blockIdx.x, tid = threadIdx.x;
  const float* x = X + (size_t)row * HH;
  float s = 0.f, s2 = 0.f;
  for (int i = tid; i < HH; i += 256) {
    float v = x[i];
    s += v;
    s2 += v * v;
  }
  sh1[tid] = s; sh2[tid] = s2;
  __syncthreads();
  for (int off = 128; off > 0; off >>= 1) {
    if (tid < off) { sh1[tid] += sh1[tid + off]; sh2[tid] += sh2[tid + off]; }
    __syncthreads();
  }
  const float mean = sh1[0] * (1.0f / HH);
  const float var  = sh2[0] * (1.0f / HH) - mean * mean;
  const float inv  = rsqrtf(var + 1e-5f);
  unsigned short* o = out + (size_t)row * HH;
  for (int i = tid; i < HH; i += 256)
    o[i] = f32_to_bf16((x[i] - mean) * inv * gam[i] + bet[i]);
}

// ---------------------------------------------------------------- WMMA GEMMs

// C[M,Nout] = A[M,K](bf16) * W[Nout,K]^T(bf16) + bias + rescale.*resid
// Block: 512 threads = 16 waves (8 m-waves x 2 n-waves); block tile 256 x 128.
// Wave tile: 32 rows x 64 cols (2 A-frags x 4 B-frags = 8 WMMAs / k-step).
// B tile (128x32) staged in LDS, double buffered. The k-loop is unrolled by 2
// with alternating register sets (no copies); next-tile global loads are issued
// AFTER each barrier so they overlap the WMMA chain, and the barriers only
// drain DScnt so those loads also fly across barriers.
// REQUIRES: M % 256 == 0, Nout % 128 == 0, K % 64 == 0 (nsteps even, >= 2).
template <bool OUT_BF16>
__global__ void __launch_bounds__(512) gemm_bf16_kernel(
    const unsigned short* __restrict__ A, int lda,
    const unsigned short* __restrict__ W, int ldw,
    const float* __restrict__ bias,
    const float* __restrict__ resid, const float* __restrict__ rescale, int ldr,
    float* __restrict__ outf, unsigned short* __restrict__ outb, int ldc,
    int M, int Nout, int K) {
  __shared__ unsigned short Bs[2][128 * BSTRIDE];   // 2 x 10KB

  const int tid   = threadIdx.x;
  const int lane  = tid & 31;
  const int wave  = tid >> 5;        // 0..15
  const int mwave = wave >> 1;       // 0..7
  const int nwave = wave & 1;        // 0..1
  const int m0 = (blockIdx.y * 8 + mwave) * 32;
  const int n0 = blockIdx.x * 128 + nwave * 64;

  // cooperative B-stage mapping: 512 threads x 8 halves = 128x32 tile
  const int brow = tid >> 2;          // 0..127  (weight row blockIdx.x*128+brow)
  const int bseg = (tid & 3) * 8;     // 0,8,16,24 (k offset)
  const unsigned short* wsrc = W + (size_t)(blockIdx.x * 128 + brow) * ldw + bseg;
  unsigned short* bdst0 = &Bs[0][brow * BSTRIDE + bseg];
  unsigned short* bdst1 = &Bs[1][brow * BSTRIDE + bseg];
  const unsigned short* bfrag0 = &Bs[0][nwave * 64 * BSTRIDE];
  const unsigned short* bfrag1 = &Bs[1][nwave * 64 * BSTRIDE];

  const int nsteps = K >> 5;          // even

  v8f acc[2][4];
#pragma unroll
  for (int mt = 0; mt < 2; ++mt)
#pragma unroll
    for (int t = 0; t < 4; ++t)
      acc[mt][t] = (v8f){0.f, 0.f, 0.f, 0.f, 0.f, 0.f, 0.f, 0.f};

  // ---- prologue: stage tile 0 and even-set A-frags for k0=0
  *(v8bf*)bdst0 = *(const v8bf*)(wsrc);
  v16bf a0e = load_a_frag(A, lda, m0,      0, lane);
  v16bf a1e = load_a_frag(A, lda, m0 + 16, 0, lane);
  v16bf a0o, a1o;

  for (int i = 0; i < nsteps; i += 2) {
    // ---- even half: compute Bs[0], load odd-step data (overlaps WMMAs)
    lds_barrier();                             // Bs[0] stores visible
    {
      const int kn = (i + 1) << 5;
      v8bf stg = *(const v8bf*)(wsrc + kn);    // issued post-barrier
      a0o = load_a_frag(A, lda, m0,      kn, lane);
      a1o = load_a_frag(A, lda, m0 + 16, kn, lane);
      mma_step(bfrag0, lane, a0e, a1e, acc);   // 8 WMMAs while loads fly
      *(v8bf*)bdst1 = stg;                     // load waited here, post-compute
    }
    // ---- odd half: compute Bs[1], load next even-step data
    lds_barrier();                             // Bs[1] stores visible
    {
      const bool more = (i + 2) < nsteps;
      v8bf stg2;
      if (more) {
        const int kn = (i + 2) << 5;
        stg2 = *(const v8bf*)(wsrc + kn);
        a0e = load_a_frag(A, lda, m0,      kn, lane);   // even set reloaded in place
        a1e = load_a_frag(A, lda, m0 + 16, kn, lane);
        __builtin_prefetch((const void*)(wsrc + ((i + 3) << 5)), 0, 1);
      }
      mma_step(bfrag1, lane, a0o, a1o, acc);
      if (more) *(v8bf*)bdst0 = stg2;
    }
  }

  // ---- epilogue
  const int half = lane >> 4, lrow = lane & 15;
#pragma unroll
  for (int mt = 0; mt < 2; ++mt) {
#pragma unroll
    for (int t = 0; t < 4; ++t) {
      const int n = n0 + t * 16 + lrow;
#pragma unroll
      for (int r = 0; r < 8; ++r) {
        const int m = m0 + mt * 16 + half * 8 + r;
        float v = acc[mt][t][r];
        if (bias) v += bias[n];
        if (resid) {
          float rr = resid[(size_t)m * ldr + n];
          v += rescale ? rescale[n] * rr : rr;
        }
        if (OUT_BF16) outb[(size_t)m * ldc + n] = f32_to_bf16(v);
        else          outf[(size_t)m * ldc + n] = v;
      }
    }
  }
}

// Fused dual GEMM: u = sigmoid(x*Wg^T + bg) .* (x*Wb^T), sharing A fragments.
// Same pipelined-LDS scheme, both weight tiles staged; 2x-unrolled k-loop.
// Block: 256 threads, tile 256 rows x 64 cols.
// REQUIRES: M % 256 == 0, N % 64 == 0, K % 64 == 0.
__global__ void __launch_bounds__(256) gate_u_kernel(
    const unsigned short* __restrict__ X,
    const unsigned short* __restrict__ Wg,
    const unsigned short* __restrict__ Wb,
    const float* __restrict__ bg,
    float* __restrict__ U, int N, int M, int K) {
  __shared__ unsigned short Gs[2][64 * BSTRIDE];    // 2 x 5KB (Wg tile)
  __shared__ unsigned short Hs[2][64 * BSTRIDE];    // 2 x 5KB (Wb tile)

  const int tid  = threadIdx.x;
  const int lane = tid & 31;
  const int wave = tid >> 5;
  const int m0 = (blockIdx.y * 8 + wave) * 32;
  const int n0 = blockIdx.x * 64;

  const int brow = tid >> 2;
  const int bseg = (tid & 3) * 8;
  const unsigned short* gsrc = Wg + (size_t)(n0 + brow) * K + bseg;
  const unsigned short* hsrc = Wb + (size_t)(n0 + brow) * K + bseg;
  unsigned short* gdst0 = &Gs[0][brow * BSTRIDE + bseg];
  unsigned short* gdst1 = &Gs[1][brow * BSTRIDE + bseg];
  unsigned short* hdst0 = &Hs[0][brow * BSTRIDE + bseg];
  unsigned short* hdst1 = &Hs[1][brow * BSTRIDE + bseg];

  const int nsteps = K >> 5;          // even

  v8f ag[2][4], ab[2][4];
#pragma unroll
  for (int mt = 0; mt < 2; ++mt)
#pragma unroll
    for (int t = 0; t < 4; ++t) {
      ag[mt][t] = (v8f){0.f, 0.f, 0.f, 0.f, 0.f, 0.f, 0.f, 0.f};
      ab[mt][t] = (v8f){0.f, 0.f, 0.f, 0.f, 0.f, 0.f, 0.f, 0.f};
    }

  // prologue
  *(v8bf*)gdst0 = *(const v8bf*)(gsrc);
  *(v8bf*)hdst0 = *(const v8bf*)(hsrc);
  v16bf a0e = load_a_frag(X, K, m0,      0, lane);
  v16bf a1e = load_a_frag(X, K, m0 + 16, 0, lane);
  v16bf a0o, a1o;

  for (int i = 0; i < nsteps; i += 2) {
    lds_barrier();
    {
      const int kn = (i + 1) << 5;
      v8bf gstg = *(const v8bf*)(gsrc + kn);
      v8bf hstg = *(const v8bf*)(hsrc + kn);
      a0o = load_a_frag(X, K, m0,      kn, lane);
      a1o = load_a_frag(X, K, m0 + 16, kn, lane);
      mma_step(Gs[0], lane, a0e, a1e, ag);
      mma_step(Hs[0], lane, a0e, a1e, ab);
      *(v8bf*)gdst1 = gstg;
      *(v8bf*)hdst1 = hstg;
    }
    lds_barrier();
    {
      const bool more = (i + 2) < nsteps;
      v8bf gstg2, hstg2;
      if (more) {
        const int kn = (i + 2) << 5;
        gstg2 = *(const v8bf*)(gsrc + kn);
        hstg2 = *(const v8bf*)(hsrc + kn);
        a0e = load_a_frag(X, K, m0,      kn, lane);
        a1e = load_a_frag(X, K, m0 + 16, kn, lane);
      }
      mma_step(Gs[1], lane, a0o, a1o, ag);
      mma_step(Hs[1], lane, a0o, a1o, ab);
      if (more) { *(v8bf*)gdst0 = gstg2; *(v8bf*)hdst0 = hstg2; }
    }
  }

  const int half = lane >> 4, lrow = lane & 15;
#pragma unroll
  for (int mt = 0; mt < 2; ++mt) {
#pragma unroll
    for (int t = 0; t < 4; ++t) {
      const int n = n0 + t * 16 + lrow;
#pragma unroll
      for (int r = 0; r < 8; ++r) {
        const int m = m0 + mt * 16 + half * 8 + r;
        float gate = 1.0f / (1.0f + expf(-(ag[mt][t][r] + bg[n])));
        U[(size_t)m * N + n] = gate * ab[mt][t][r];
      }
    }
  }
}

// ---------------------------------------------------------------- launch

extern "C" void kernel_launch(void* const* d_in, const int* in_sizes, int n_in,
                              void* d_out, int out_size, void* d_ws, size_t ws_size,
                              hipStream_t stream) {
  (void)in_sizes; (void)n_in; (void)out_size; (void)ws_size;
  const int SDSa[3] = {64, 128, 256};

  const int*   xids = (const int*)d_in[0];
  const float* E    = (const float*)d_in[1];
  const float *Alog[3], *Wb[3], *Wc[3], *Dv[3], *Wg[3], *bg[3], *Wo[3], *bo[3], *gm[3], *be[3];
  for (int i = 0; i < 3; ++i) {
    const int b = 2 + 10 * i;
    Alog[i] = (const float*)d_in[b + 0];
    Wb[i]   = (const float*)d_in[b + 1];
    Wc[i]   = (const float*)d_in[b + 2];
    Dv[i]   = (const float*)d_in[b + 3];
    Wg[i]   = (const float*)d_in[b + 4];
    bg[i]   = (const float*)d_in[b + 5];
    Wo[i]   = (const float*)d_in[b + 6];
    bo[i]   = (const float*)d_in[b + 7];
    gm[i]   = (const float*)d_in[b + 8];
    be[i]   = (const float*)d_in[b + 9];
  }
  const float* Wf  = (const float*)d_in[32];
  const float* bf  = (const float*)d_in[33];
  const float* gf  = (const float*)d_in[34];
  const float* bef = (const float*)d_in[35];
  const float* Wh  = (const float*)d_in[36];
  const float* bh  = (const float*)d_in[37];

  // ---- workspace carve (256B aligned slabs)
  char* ws = (char*)d_ws;
  size_t cur = 0;
  auto alloc = [&](size_t bytes) -> void* {
    void* p = ws + cur;
    cur = (cur + bytes + 255) & ~(size_t)255;
    return p;
  };
  float*          emb_f = (float*)alloc((size_t)MM * HH * 4);
  unsigned short* emb_b = (unsigned short*)alloc((size_t)MM * HH * 2);
  float*          U     = (float*)alloc((size_t)MM * 256 * 4);
  unsigned short* St    = (unsigned short*)alloc((size_t)MM * 256 * 2);
  float*          T     = (float*)alloc((size_t)MM * HH * 4);          // y-resid / fused
  unsigned short* Z     = (unsigned short*)alloc((size_t)MM * HH * 2); // LN out (bf16)
  unsigned short* Comb  = (unsigned short*)alloc((size_t)MM * 3 * HH * 2);
  unsigned short *Wgb[3], *Wbb[3], *Wcb[3], *Wob[3];
  float* dp1[3];
  for (int i = 0; i < 3; ++i) {
    Wgb[i] = (unsigned short*)alloc((size_t)SDSa[i] * HH * 2);
    Wbb[i] = (unsigned short*)alloc((size_t)SDSa[i] * HH * 2);
    Wcb[i] = (unsigned short*)alloc((size_t)HH * SDSa[i] * 2);
    Wob[i] = (unsigned short*)alloc((size_t)HH * HH * 2);
    dp1[i] = (float*)alloc((size_t)HH * 4);
  }
  unsigned short* Wfb = (unsigned short*)alloc((size_t)HH * 3 * HH * 2);
  unsigned short* Whb = (unsigned short*)alloc((size_t)VV * HH * 2);

  auto cvt = [&](const float* in, unsigned short* out, size_t n) {
    cvt_kernel<<<(unsigned)((n + 255) / 256), 256, 0, stream>>>(in, out, n);
  };

  // ---- weight prep
  for (int i = 0; i < 3; ++i) {
    cvt(Wg[i], Wgb[i], (size_t)SDSa[i] * HH);
    cvt(Wb[i], Wbb[i], (size_t)SDSa[i] * HH);
    cvt(Wc[i], Wcb[i], (size_t)HH * SDSa[i]);
    cvt(Wo[i], Wob[i], (size_t)HH * HH);
    dplus1_kernel<<<(HH + 255) / 256, 256, 0, stream>>>(Dv[i], dp1[i], HH);
  }
  cvt(Wf, Wfb, (size_t)HH * 3 * HH);
  cvt(Wh, Whb, (size_t)VV * HH);

  // ---- embedding gather
  {
    size_t tot = (size_t)MM * HH;
    embed_kernel<<<(unsigned)((tot + 255) / 256), 256, 0, stream>>>(xids, E, emb_f, emb_b, tot);
  }

  // ---- SSM layers
  for (int i = 0; i < 3; ++i) {
    const int N = SDSa[i];
    dim3 gu(N / 64, MM / 256);
    gate_u_kernel<<<gu, 256, 0, stream>>>(emb_b, Wgb[i], Wbb[i], bg[i], U, N, MM, HH);

    int tot = BB * N;
    scan_kernel<<<(tot + 255) / 256, 256, 0, stream>>>(U, Alog[i], St, N);

    // T = states*Wc^T + (D+1).*emb     (bias=null, resid=emb, rescale=D+1)
    dim3 gc(HH / 128, MM / 256);
    gemm_bf16_kernel<false><<<gc, 512, 0, stream>>>(
        St, N, Wcb[i], N, nullptr, emb_f, dp1[i], HH, T, nullptr, HH, MM, HH, N);

    ln_kernel<<<MM, 256, 0, stream>>>(T, gm[i], be[i], Z);

    // Comb[:, i*H:(i+1)*H] = Z*Wo^T + bo   (bf16 output)
    dim3 go(HH / 128, MM / 256);
    gemm_bf16_kernel<true><<<go, 512, 0, stream>>>(
        Z, HH, Wob[i], HH, bo[i], nullptr, nullptr, 0,
        nullptr, Comb + (size_t)i * HH, 3 * HH, MM, HH, HH);
  }

  // ---- fuse: T = Comb*Wf^T + bf + emb ; Z = LN(T)
  {
    dim3 gfd(HH / 128, MM / 256);
    gemm_bf16_kernel<false><<<gfd, 512, 0, stream>>>(
        Comb, 3 * HH, Wfb, 3 * HH, bf, emb_f, nullptr, HH, T, nullptr, HH, MM, HH, 3 * HH);
    ln_kernel<<<MM, 256, 0, stream>>>(T, gf, bef, Z);
  }

  // ---- head: logits = Z*Wh^T + bh   (dominant GEMM: 4096x32000x1024)
  {
    dim3 gh(VV / 128, MM / 256);
    gemm_bf16_kernel<false><<<gh, 512, 0, stream>>>(
        Z, HH, Whb, HH, bh, nullptr, nullptr, 0,
        (float*)d_out, nullptr, VV, MM, VV, HH);
  }
}
